// CausalSelfAttention_9251359555858
// MI455X (gfx1250) — compile-verified, matrix operand order checked by reading, not run on hardware
//
#include <hip/hip_runtime.h>

// ---------------------------------------------------------------------------
// Problem constants (fixed by the reference)
// ---------------------------------------------------------------------------
#define BSZ   4
#define TLEN  2048
#define CDIM  2048
#define NH    16
#define DH    128
#define NQKV  (3 * CDIM)
#define MROWS (BSZ * TLEN)          // 8192

typedef __bf16 bf16_t;
typedef __attribute__((ext_vector_type(4)))  __bf16 v4bf;
typedef __attribute__((ext_vector_type(8)))  __bf16 v8bf;
typedef __attribute__((ext_vector_type(16))) __bf16 v16bf;
typedef __attribute__((ext_vector_type(8)))  float  v8f;

// D(16x16,f32) = A(16x32,bf16) * B(32x16,bf16) + C
__device__ __forceinline__ v8f wmma_bf16(v16bf a, v16bf b, v8f c) {
  return __builtin_amdgcn_wmma_f32_16x16x32_bf16(
      /*neg_a=*/false, a, /*neg_b=*/false, b,
      /*c_mod=*/(short)0, c, /*reuse_a=*/false, /*reuse_b=*/false);
}

// One WMMA operand fragment (16x32 bf16). Lane l holds row (l&15); half
// (l>>4) owns contiguous K chunks kh*8..kh*8+7 (VGPR0-3) and +16 (VGPR4-7).
// `row` points at a 32-element contraction-major row (global or LDS).
__device__ __forceinline__ v16bf load_frag(const bf16_t* row, int kh) {
  v8bf lo = *(const v8bf*)(row + kh * 8);
  v8bf hi = *(const v8bf*)(row + 16 + kh * 8);
  return __builtin_shufflevector(lo, hi, 0, 1, 2, 3, 4, 5, 6, 7,
                                 8, 9, 10, 11, 12, 13, 14, 15);
}

// Row reductions across the 16-lane half holding one C-fragment row.
__device__ __forceinline__ float rowmax16(float v) {
  #pragma unroll
  for (int m = 1; m < 16; m <<= 1) v = fmaxf(v, __shfl_xor(v, m, 32));
  return v;
}
__device__ __forceinline__ float rowsum16(float v) {
  #pragma unroll
  for (int m = 1; m < 16; m <<= 1) v += __shfl_xor(v, m, 32);
  return v;
}

// ---------------------------------------------------------------------------
// fp32 -> bf16 conversion (x activations)
// ---------------------------------------------------------------------------
__global__ __launch_bounds__(256) void cvt_f32_to_bf16(
    const float* __restrict__ in, bf16_t* __restrict__ out) {
  size_t i = ((size_t)blockIdx.x * 256 + threadIdx.x) * 4;
  float4 v = *(const float4*)(in + i);
  v4bf o;
  o[0] = (bf16_t)v.x; o[1] = (bf16_t)v.y; o[2] = (bf16_t)v.z; o[3] = (bf16_t)v.w;
  *(v4bf*)(out + i) = o;
}

// ---------------------------------------------------------------------------
// One-time weight prep: W[K][Nn] fp32 -> Wt[Nn][K=2048] bf16 (transposed).
// 32x32 tiles through padded LDS; both global accesses coalesced.
// ---------------------------------------------------------------------------
__global__ __launch_bounds__(256) void transpose_f32_to_bf16(
    const float* __restrict__ W, bf16_t* __restrict__ Wt, int Nn) {
  __shared__ float tile[32][33];
  const int n0 = blockIdx.x * 32, k0 = blockIdx.y * 32;
  const int tx = threadIdx.x & 31, ty = threadIdx.x >> 5;  // 32 x 8
  #pragma unroll
  for (int i = 0; i < 4; ++i) {
    int r = ty + i * 8;
    tile[r][tx] = W[(size_t)(k0 + r) * Nn + n0 + tx];
  }
  __syncthreads();
  #pragma unroll
  for (int i = 0; i < 4; ++i) {
    int r = ty + i * 8;
    Wt[(size_t)(n0 + r) * CDIM + k0 + tx] = (bf16_t)tile[tx][r];
  }
}

// ---------------------------------------------------------------------------
// WMMA GEMM: D[M,N] = A_bf16[M,2048] * Bt_bf16[N,2048]^T + bias
// Block = 256 threads (8 waves), tile 128(M) x 128(N), K-step 32.
// Waves 2(M) x 4(N); wave tile 64x32 = 4x2 WMMA tiles (8 WMMA / 6 frag loads).
// QKV=true : scatter bf16 into Q[B,H,T,D], K[B,H,T,D], V^T[B,H,D,T]
// QKV=false: fp32 + bias into `out` (final projection)
// ---------------------------------------------------------------------------
template <bool QKV>
__global__ __launch_bounds__(256) void gemm_bf16_kernel(
    const bf16_t* __restrict__ A, const bf16_t* __restrict__ Bt,
    const float* __restrict__ bias,
    bf16_t* __restrict__ q_ws, bf16_t* __restrict__ k_ws,
    bf16_t* __restrict__ vT_ws, float* __restrict__ out) {
  __shared__ __attribute__((aligned(16))) bf16_t As[128][32];
  __shared__ __attribute__((aligned(16))) bf16_t Bs[128][32];

  const int tid = threadIdx.x;
  const int w = tid >> 5, lane = tid & 31;
  const int wm = w >> 2, wn = w & 3;          // 2 x 4 wave grid
  const int lr = lane & 15, kh = lane >> 4;
  const int mBase = blockIdx.y * 128;
  const int nBase = blockIdx.x * 128;

  // Running global pointers for this thread's two staging rows.
  const int rA0 = tid >> 2, cvA = (tid & 3) * 8;
  const bf16_t* gA0 = A + (size_t)(mBase + rA0) * CDIM + cvA;
  const bf16_t* gA1 = A + (size_t)(mBase + rA0 + 64) * CDIM + cvA;
  const bf16_t* gB0 = Bt + (size_t)(nBase + rA0) * CDIM + cvA;
  const bf16_t* gB1 = Bt + (size_t)(nBase + rA0 + 64) * CDIM + cvA;

  v8f acc[4][2] = {};

  for (int k0 = 0; k0 < CDIM; k0 += 32) {
    // Prefetch next K tile (global_prefetch_b8, CDNA5).
    if (k0 + 32 < CDIM) {
      __builtin_prefetch(gA0 + 32, 0, 1);
      __builtin_prefetch(gB0 + 32, 0, 1);
    }
    // Stage A and B tiles: 2 x B128 copies each per thread.
    *(v8bf*)&As[rA0][cvA]      = *(const v8bf*)gA0;
    *(v8bf*)&As[rA0 + 64][cvA] = *(const v8bf*)gA1;
    *(v8bf*)&Bs[rA0][cvA]      = *(const v8bf*)gB0;
    *(v8bf*)&Bs[rA0 + 64][cvA] = *(const v8bf*)gB1;
    gA0 += 32; gA1 += 32; gB0 += 32; gB1 += 32;
    __syncthreads();

    v16bf bfr[2], afr[4];
    #pragma unroll
    for (int j = 0; j < 2; ++j)
      bfr[j] = load_frag(&Bs[wn * 32 + j * 16 + lr][0], kh);
    #pragma unroll
    for (int i = 0; i < 4; ++i)
      afr[i] = load_frag(&As[wm * 64 + i * 16 + lr][0], kh);
    #pragma unroll
    for (int i = 0; i < 4; ++i)
      #pragma unroll
      for (int j = 0; j < 2; ++j)
        acc[i][j] = wmma_bf16(afr[i], bfr[j], acc[i][j]);
    __syncthreads();
  }

  // Epilogue. C-fragment: lane holds col N=lr; VGPR r holds row r + 8*kh.
  #pragma unroll
  for (int i = 0; i < 4; ++i) {
    #pragma unroll
    for (int j = 0; j < 2; ++j) {
      const int gCol = nBase + wn * 32 + j * 16 + lr;
      const float bv = bias[gCol];
      #pragma unroll
      for (int r = 0; r < 8; ++r) {
        const int gRow = mBase + wm * 64 + i * 16 + r + 8 * kh;
        const float val = acc[i][j][r] + bv;
        if (QKV) {
          int which = gCol >> 11;   // 0=q 1=k 2=v
          int c = gCol & (CDIM - 1);
          int hh = c >> 7, d = c & (DH - 1);
          int bb = gRow >> 11, t = gRow & (TLEN - 1);
          if (which == 0)
            q_ws[(((size_t)(bb * NH + hh)) * TLEN + t) * DH + d] = (bf16_t)val;
          else if (which == 1)
            k_ws[(((size_t)(bb * NH + hh)) * TLEN + t) * DH + d] = (bf16_t)val;
          else  // V stored transposed: [B,H,D,T] so P*V B-frags are contiguous
            vT_ws[(((size_t)(bb * NH + hh)) * DH + d) * TLEN + t] = (bf16_t)val;
        } else {
          out[(size_t)gRow * CDIM + gCol] = val;
        }
      }
    }
  }
}

// ---------------------------------------------------------------------------
// Flash-style causal attention: one wave per 16-query tile of one (b,h).
// Q in registers (4 A-fragments). Per 32-key chunk: preload all 8 K-frags
// and all 8 V^T-frags into distinct registers (lets the scheduler issue one
// big load clause and use partial s_wait_loadcnt), then S = Q*K^T (8 WMMAs),
// online softmax (overlapped with the V^T loads in flight), O += P*V
// (8 WMMAs). Only P passes through LDS (C-layout -> A-layout).
// ---------------------------------------------------------------------------
__global__ __launch_bounds__(32) void attn_kernel(
    const bf16_t* __restrict__ q_ws, const bf16_t* __restrict__ k_ws,
    const bf16_t* __restrict__ vT_ws, bf16_t* __restrict__ y_bf) {
  __shared__ __attribute__((aligned(16))) bf16_t Pb[16][32];

  const int lane = threadIdx.x;
  const int lr = lane & 15, kh = lane >> 4;
  const int qt = blockIdx.x & (TLEN / 16 - 1);
  const int hb = blockIdx.x / (TLEN / 16);
  const int h = hb & (NH - 1), b = hb >> 4;
  const size_t head_off = ((size_t)(b * NH + h)) * TLEN * DH;  // == H*D*T slice
  const bf16_t* Q  = q_ws  + head_off;
  const bf16_t* K  = k_ws  + head_off;
  const bf16_t* VT = vT_ws + head_off;   // [D][T] rows
  const int qbase = qt * 16;

  v16bf qf[4];
  #pragma unroll
  for (int kk = 0; kk < 4; ++kk)
    qf[kk] = load_frag(Q + (size_t)(qbase + lr) * DH + kk * 32, kh);

  float mrow[8], lrow[8];
  v8f O[8] = {};
  #pragma unroll
  for (int r = 0; r < 8; ++r) { mrow[r] = -1e30f; lrow[r] = 0.f; }

  const float scale = 0.08838834764831845f;  // 1/sqrt(128)

  // Running per-lane base pointers; advanced by one 32-key chunk per iter so
  // the inner loop uses pure immediate-offset addressing.
  const bf16_t* kp = K  + (size_t)lr * DH;    // += 32*DH per chunk
  const bf16_t* vp = VT + (size_t)lr * TLEN;  // += 32 per chunk

  for (int kc = 0; kc <= qbase + 15; kc += 32) {
    if (kc + 32 <= qbase + 15)  // prefetch next chunk's K rows
      __builtin_prefetch(kp + 32 * DH, 0, 1);

    // Preload the whole chunk: 8 K fragments + 8 V^T fragments.
    v16bf kfr[2][4], vfr[8];
    #pragma unroll
    for (int j = 0; j < 2; ++j)
      #pragma unroll
      for (int kk = 0; kk < 4; ++kk)
        kfr[j][kk] = load_frag(kp + j * 16 * DH + kk * 32, kh);
    #pragma unroll
    for (int f = 0; f < 8; ++f)
      vfr[f] = load_frag(vp + (size_t)f * 16 * TLEN, kh);

    // S = Q * K^T for two 16-key sub-tiles.
    v8f s[2];
    #pragma unroll
    for (int j = 0; j < 2; ++j) {
      v8f a = {};
      #pragma unroll
      for (int kk = 0; kk < 4; ++kk)
        a = wmma_bf16(qf[kk], kfr[j][kk], a);
      s[j] = a;
    }

    // Scale + causal mask (lane col = kc+j*16+lr, row = qbase+r+8*kh).
    #pragma unroll
    for (int j = 0; j < 2; ++j) {
      #pragma unroll
      for (int r = 0; r < 8; ++r) {
        int qg = qbase + r + 8 * kh;
        int key = kc + j * 16 + lr;
        s[j][r] = (key <= qg) ? s[j][r] * scale : -1e30f;
      }
    }

    // Online softmax: row max/sum via 16-lane shuffles; P -> LDS as bf16.
    float alpha[8];
    #pragma unroll
    for (int r = 0; r < 8; ++r) {
      float mx = rowmax16(fmaxf(s[0][r], s[1][r]));
      float mnew = fmaxf(mrow[r], mx);
      float a = __expf(mrow[r] - mnew);
      float p0 = __expf(s[0][r] - mnew);
      float p1 = __expf(s[1][r] - mnew);
      lrow[r] = lrow[r] * a + rowsum16(p0 + p1);
      mrow[r] = mnew;
      alpha[r] = a;
      Pb[r + 8 * kh][lr] = (bf16_t)p0;       // C-layout -> A-layout via LDS
      Pb[r + 8 * kh][16 + lr] = (bf16_t)p1;
    }
    #pragma unroll
    for (int f = 0; f < 8; ++f)
      #pragma unroll
      for (int r = 0; r < 8; ++r) O[f][r] *= alpha[r];

    __syncthreads();

    // O += P(16x32) * V(32x128).
    v16bf pf = load_frag(&Pb[lr][0], kh);
    #pragma unroll
    for (int f = 0; f < 8; ++f)
      O[f] = wmma_bf16(pf, vfr[f], O[f]);

    __syncthreads();  // Pb reused next chunk

    kp += 32 * DH;
    vp += 32;
  }

  // Normalize, write y bf16 in [B,T,C] layout (col = h*DH + d).
  #pragma unroll
  for (int f = 0; f < 8; ++f) {
    #pragma unroll
    for (int r = 0; r < 8; ++r) {
      int t = qbase + r + 8 * kh;
      int col = h * DH + f * 16 + lr;
      y_bf[((size_t)(b * TLEN + t)) * CDIM + col] = (bf16_t)(O[f][r] / lrow[r]);
    }
  }
}

// ---------------------------------------------------------------------------
// Host-side orchestration
// ---------------------------------------------------------------------------
extern "C" void kernel_launch(void* const* d_in, const int* in_sizes, int n_in,
                              void* d_out, int out_size, void* d_ws,
                              size_t ws_size, hipStream_t stream) {
  (void)in_sizes; (void)n_in; (void)out_size; (void)ws_size;
  const float* x      = (const float*)d_in[0];
  const float* w_attn = (const float*)d_in[1];
  const float* b_attn = (const float*)d_in[2];
  const float* w_proj = (const float*)d_in[3];
  const float* b_proj = (const float*)d_in[4];
  float* out = (float*)d_out;

  // Workspace (bf16 elements): x | q | k | vT | y | wattn^T | wproj^T  ~201MB
  const size_t BTC = (size_t)BSZ * TLEN * CDIM;  // 16,777,216
  bf16_t* x_bf    = (bf16_t*)d_ws;
  bf16_t* q_ws    = x_bf + BTC;
  bf16_t* k_ws    = q_ws + BTC;
  bf16_t* vT_ws   = k_ws + BTC;
  bf16_t* y_bf    = vT_ws + BTC;
  bf16_t* wattn_t = y_bf + BTC;                       // [6144][2048]
  bf16_t* wproj_t = wattn_t + (size_t)NQKV * CDIM;    // [2048][2048]

  // 1) x -> bf16; weights -> bf16 transposed (one-time prep)
  cvt_f32_to_bf16<<<(unsigned)(BTC / (4 * 256)), 256, 0, stream>>>(x, x_bf);
  transpose_f32_to_bf16<<<dim3(NQKV / 32, CDIM / 32), 256, 0, stream>>>(
      w_attn, wattn_t, NQKV);
  transpose_f32_to_bf16<<<dim3(CDIM / 32, CDIM / 32), 256, 0, stream>>>(
      w_proj, wproj_t, CDIM);

  // 2) QKV GEMM -> Q[B,H,T,D], K[B,H,T,D], V^T[B,H,D,T] (bf16)
  gemm_bf16_kernel<true>
      <<<dim3(NQKV / 128, MROWS / 128), 256, 0, stream>>>(
          x_bf, wattn_t, b_attn, q_ws, k_ws, vT_ws, nullptr);

  // 3) Causal flash attention: one wave per (b,h,16-query tile)
  attn_kernel<<<dim3(BSZ * NH * (TLEN / 16)), 32, 0, stream>>>(
      q_ws, k_ws, vT_ws, y_bf);

  // 4) Output projection -> fp32 out
  gemm_bf16_kernel<false>
      <<<dim3(CDIM / 128, MROWS / 128), 256, 0, stream>>>(
          y_bf, wproj_t, b_proj, nullptr, nullptr, nullptr, out);
}
